// ConvNet_19396072309309
// MI455X (gfx1250) — matrix operand order repaired and checked by reading, not compiled, not run on hardware
//
#include <hip/hip_runtime.h>
#include <hip/hip_bf16.h>

typedef __attribute__((ext_vector_type(16))) _Float16 v16h;
typedef __attribute__((ext_vector_type(8)))  _Float16 v8h;
typedef __attribute__((ext_vector_type(4)))  _Float16 v4h;
typedef __attribute__((ext_vector_type(8)))  float    v8f;

#define HW    4096
#define CCH   256
#define PP    8
#define BB    32
#define ROWH  264   // xh/hh LDS row stride: 256 + 8-half pad (16B aligned rows)
#define MROW  40    // maskT/xT row stride: 32 pix + 8-half pad (16B aligned rows)
#define NPBLK 16    // pixel-blocks per batch (256 pixels each)

// ---------------------------------------------------------------------------
// prep: fold BN into W1, convert weights to f16, zero-pad W2 to 16 rows
// ---------------------------------------------------------------------------
__global__ __launch_bounds__(256) void prep_kernel(
    const float* __restrict__ W1, const float* __restrict__ b1,
    const float* __restrict__ gamma, const float* __restrict__ beta,
    const float* __restrict__ mean, const float* __restrict__ var,
    const float* __restrict__ W2,
    _Float16* __restrict__ W1h, _Float16* __restrict__ W2h,
    float* __restrict__ b1f) {
  int blk = blockIdx.x, t = threadIdx.x;
  if (blk < 256) {
    int o = blk;
    float inv = gamma[o] * rsqrtf(var[o] + 1e-5f);
    W1h[o * CCH + t] = (_Float16)(W1[o * CCH + t] * inv);
    if (t == 0) b1f[o] = b1[o] * inv + beta[o] - mean[o] * inv;
  } else {
    for (int i = 0; i < 16; ++i) {
      int idx = i * 256 + t;
      int row = idx >> 8, col = idx & 255;
      W2h[idx] = (row < PP) ? (_Float16)W2[row * CCH + col] : (_Float16)0.0f;
    }
  }
}

// ---------------------------------------------------------------------------
// fused: per 256-pixel block (8 sub-tiles of 32 pixels), all in one x pass:
//   h = relu(W1f*x + b1f)                (GEMM1, 16Mx2N WMMA tiles)
//   mask = sigmoid(W2*h + b2)            (GEMM2, 1 WMMA tile x 2N)
//   vec_partial += x_tile * maskT        (GEMM3, K=32 WMMA per c-tile)
// x is read from HBM exactly once. Deterministic: no atomics.
// ---------------------------------------------------------------------------
__global__ __launch_bounds__(256) void fused_kernel(
    const float* __restrict__ x, const _Float16* __restrict__ W1h,
    const float* __restrict__ b1f, const _Float16* __restrict__ W2h,
    const float* __restrict__ b2,
    float* __restrict__ partial,      // [512][C*P] f32 per-block vec partials
    float* __restrict__ psumw) {      // [512][2][8] per-block mask sums
  __shared__ _Float16 xh[32 * ROWH];      // [pix][c]   f16 (GEMM1 B operand)
  __shared__ _Float16 hh[32 * ROWH];      // [pix][o]   f16 (GEMM2 B operand)
  __shared__ _Float16 xT[CCH * MROW];     // [c][pix]   f16 (GEMM3 A operand)
  __shared__ _Float16 maskT[16 * MROW];   // [p16][pix] f16 (GEMM3 B operand)

  int b    = blockIdx.x >> 4;
  int hwbl = (blockIdx.x & (NPBLK - 1)) * 256;
  int t = threadIdx.x, w = t >> 5, lane = t & 31;
  int ln15 = lane & 15, hi = lane >> 4;
  int klo  = hi ? 16 : 0;      // f16 A-fragment K sub-offset
  int koff = hi ? 16 : 0;      // f16 B-fragment K sub-offset

  // zero maskT once (rows 8..15 stay zero forever; pad cols never read)
  for (int i = t; i < 16 * MROW; i += 256) maskT[i] = (_Float16)0.0f;

  v8f accV[2] = {{}, {}};      // vec accumulators: c-tiles w and w+8
  float ps[8];                 // per-lane partial mask sums (waves 0,1)
  for (int r = 0; r < 8; ++r) ps[r] = 0.0f;

  for (int s5 = 0; s5 < 8; ++s5) {
    int hw0 = hwbl + s5 * 32;
    __syncthreads();  // previous iteration's readers of xh/xT/maskT are done

    { // stage x sub-tile -> f16 LDS, both layouts.
      // float4 loads: 8 lanes x 16B = 128B coalesced per c-row, clause-able.
      int j4 = (t & 7) * 4;    // pixel group
      int c0 = t >> 3;         // c row, stride 32
      const float* xp = x + (size_t)b * CCH * HW + hw0 + j4;
#pragma unroll
      for (int c = c0; c < CCH; c += 32) {
        float4 f = *(const float4*)(xp + (size_t)c * HW);
        v4h h4;
        h4[0] = (_Float16)f.x; h4[1] = (_Float16)f.y;
        h4[2] = (_Float16)f.z; h4[3] = (_Float16)f.w;
        *(v4h*)(xT + c * MROW + j4) = h4;       // packed 8B store
        xh[(j4 + 0) * ROWH + c] = h4[0];
        xh[(j4 + 1) * ROWH + c] = h4[1];
        xh[(j4 + 2) * ROWH + c] = h4[2];
        xh[(j4 + 3) * ROWH + c] = h4[3];
      }
    }
    __syncthreads();

    // ---- GEMM1: 16 M-tiles x 2 N-tiles, 4 per wave ----
#pragma unroll
    for (int i = 0; i < 4; ++i) {
      int tid = w * 4 + i;
      int m0 = (tid >> 1) * 16;
      int ntile = tid & 1;
      v8f acc = {};
#pragma unroll
      for (int k0 = 0; k0 < CCH; k0 += 32) {
        const v8h* ap = (const v8h*)(W1h + (size_t)(m0 + ln15) * CCH + k0 + klo);
        v8h a0 = ap[0], a1 = ap[1];
        v16h af;
        for (int e = 0; e < 8; ++e) { af[e] = a0[e]; af[e + 8] = a1[e]; }
        v16h bf = *(const v16h*)(xh + (ntile * 16 + ln15) * ROWH + k0 + koff);
        acc = __builtin_amdgcn_wmma_f32_16x16x32_f16(false, af, false, bf,
                                                     (short)0, acc, false, false);
      }
      int obase = m0 + 8 * hi;
      v8h hv;
      for (int r = 0; r < 8; ++r) {
        float v = acc[r] + b1f[obase + r];
        hv[r] = (_Float16)(v > 0.0f ? v : 0.0f);
      }
      *(v8h*)(hh + (ntile * 16 + ln15) * ROWH + obase) = hv;
    }
    __syncthreads();

    // ---- GEMM2 + sigmoid -> maskT (waves 0,1; wave-uniform branch) ----
    if (w < 2) {
      int ntile = w;
      v8f acc = {};
#pragma unroll
      for (int k0 = 0; k0 < CCH; k0 += 32) {
        const v8h* ap = (const v8h*)(W2h + (size_t)ln15 * CCH + k0 + klo);
        v8h a0 = ap[0], a1 = ap[1];
        v16h af;
        for (int e = 0; e < 8; ++e) { af[e] = a0[e]; af[e + 8] = a1[e]; }
        v16h bf = *(const v16h*)(hh + (ntile * 16 + ln15) * ROWH + k0 + koff);
        acc = __builtin_amdgcn_wmma_f32_16x16x32_f16(false, af, false, bf,
                                                     (short)0, acc, false, false);
      }
      if (hi == 0) {  // lanes 0..15: rows p=0..7, cols = 16 pixels of ntile
        for (int r = 0; r < PP; ++r) {
          float v = acc[r] + b2[r];
          float sg = 1.0f / (1.0f + __expf(-v));
          maskT[r * MROW + ntile * 16 + ln15] = (_Float16)sg;
          ps[r] += sg;   // f32 running sum (pre-f16 rounding, like reference)
        }
      }
    }
    __syncthreads();

    // ---- GEMM3: vec partials, 2 c-tiles per wave, single K=32 step ----
    {
      v16h bf = *(const v16h*)(maskT + ln15 * MROW + koff);  // N=p, K=pixels
#pragma unroll
      for (int m = 0; m < 2; ++m) {
        int m0 = (w + 8 * m) * 16;
        const v8h* ap = (const v8h*)(xT + (size_t)(m0 + ln15) * MROW + klo);
        v8h a0 = ap[0], a1 = ap[1];   // pixels klo+0..7 and klo+8..15
        v16h af;
        for (int e = 0; e < 8; ++e) { af[e] = a0[e]; af[e + 8] = a1[e]; }
        accV[m] = __builtin_amdgcn_wmma_f32_16x16x32_f16(false, af, false, bf,
                                                         (short)0, accV[m],
                                                         false, false);
      }
    }
  }

  // ---- write per-block vec partials (cols p=0..7 valid) ----
  if (ln15 < PP) {
    for (int m = 0; m < 2; ++m) {
      int m0 = (w + 8 * m) * 16;
      for (int r = 0; r < 8; ++r)
        partial[(size_t)blockIdx.x * (CCH * PP) +
                (size_t)(m0 + r + 8 * hi) * PP + ln15] = accV[m][r];
    }
  }

  // ---- write per-block mask sums (waves 0,1 hold ntile 0,1) ----
  if (w < 2 && hi == 0) {
    for (int r = 0; r < PP; ++r) {
      float v = ps[r];
      v += __shfl_xor(v, 1, 32);
      v += __shfl_xor(v, 2, 32);
      v += __shfl_xor(v, 4, 32);
      v += __shfl_xor(v, 8, 32);
      if (ln15 == 0) psumw[(size_t)blockIdx.x * 16 + w * 8 + r] = v;
    }
  }
}

// ---------------------------------------------------------------------------
// sumw: fixed-order reduction of per-block mask sums -> sumw[b][p]
// ---------------------------------------------------------------------------
__global__ __launch_bounds__(256) void sumw_kernel(
    const float* __restrict__ psumw, float* __restrict__ sumw) {
  int bp = threadIdx.x;            // 256 = B*P
  int b = bp >> 3, p = bp & 7;
  float s = 0.0f;
  for (int tile = 0; tile < NPBLK; ++tile) {
    size_t base = (size_t)(b * NPBLK + tile) * 16;
    s += psumw[base + p] + psumw[base + 8 + p];
  }
  sumw[bp] = s;
}

// ---------------------------------------------------------------------------
// final: fixed-order reduction of vec partials, divide by sumw, write out
// out flat = (B,C,P) row-major (reference's reshape is a raw reshape)
// ---------------------------------------------------------------------------
__global__ __launch_bounds__(256) void final_kernel(
    const float* __restrict__ partial, const float* __restrict__ sumw,
    float* __restrict__ out) {
  int idx = blockIdx.x * 256 + threadIdx.x;   // 65536 outputs
  int b  = idx >> 11;
  int cp = idx & 2047;
  int p  = cp & 7;
  float s = 0.0f;
  for (int tile = 0; tile < NPBLK; ++tile)
    s += partial[(size_t)(b * NPBLK + tile) * (CCH * PP) + cp];
  out[idx] = s / (sumw[b * PP + p] + 1e-12f);
}

// ---------------------------------------------------------------------------
extern "C" void kernel_launch(void* const* d_in, const int* in_sizes, int n_in,
                              void* d_out, int out_size, void* d_ws, size_t ws_size,
                              hipStream_t stream) {
  const float* x     = (const float*)d_in[0];
  const float* W1    = (const float*)d_in[1];
  const float* b1    = (const float*)d_in[2];
  const float* gamma = (const float*)d_in[3];
  const float* beta  = (const float*)d_in[4];
  const float* mean  = (const float*)d_in[5];
  const float* var   = (const float*)d_in[6];
  const float* W2    = (const float*)d_in[7];
  const float* b2    = (const float*)d_in[8];
  float* out = (float*)d_out;

  char* ws = (char*)d_ws;
  _Float16* W1h     = (_Float16*)(ws + 0);        // 131072 B
  _Float16* W2h     = (_Float16*)(ws + 131072);   //   8192 B
  float*    b1f     = (float*)(ws + 139264);      //   1024 B
  float*    psumw   = (float*)(ws + 140288);      // 512*16*4 = 32768 B
  float*    sumw    = (float*)(ws + 173056);      //   1024 B
  float*    partial = (float*)(ws + 174080);      // 512*2048*4 = 4 MiB

  prep_kernel<<<257, 256, 0, stream>>>(W1, b1, gamma, beta, mean, var, W2,
                                       W1h, W2h, b1f);
  fused_kernel<<<BB * NPBLK, 256, 0, stream>>>(x, W1h, b1f, W2h, b2,
                                               partial, psumw);
  sumw_kernel<<<1, 256, 0, stream>>>(psumw, sumw);
  final_kernel<<<256, 256, 0, stream>>>(partial, sumw, out);
}
